// RNNdoubleheadPred_31018253811770
// MI455X (gfx1250) — matrix operand order repaired and checked by reading, not compiled, not run on hardware
//
#include <hip/hip_runtime.h>
#include <hip/hip_bf16.h>

#define B_SZ   128
#define CHUNKS 2
#define INS    512
#define H      256
#define FOURH  1024
#define T_STEPS 1024
#define H2     512
#define OS     64
#define KBIG   (INS*H)   // 131072

typedef __attribute__((ext_vector_type(16))) __bf16 v16bf;
typedef __attribute__((ext_vector_type(8)))  float  v8f;

union FragBF { v16bf v; unsigned u[8]; };

__device__ __forceinline__ unsigned short f2bf(float f) {
  union { float f; unsigned u; } x; x.f = f;
  unsigned r = x.u + 0x7FFFu + ((x.u >> 16) & 1u);
  return (unsigned short)(r >> 16);
}
__device__ __forceinline__ unsigned pack2bf(float a, float b) {
  return (unsigned)f2bf(a) | ((unsigned)f2bf(b) << 16);
}
// sigmoid via v_exp + v_rcp (no libm call)
__device__ __forceinline__ float fast_sigmoid(float x) {
  return __builtin_amdgcn_rcpf(1.0f + __expf(-x));
}
// tanh via exp/rcp, sign-safe for large |x| (avoids __ocml_tanh_f32 call)
__device__ __forceinline__ float fast_tanh(float x) {
  float ax = fabsf(x);
  float e  = __expf(-2.0f * ax);
  float t  = (1.0f - e) * __builtin_amdgcn_rcpf(1.0f + e);
  return copysignf(t, x);
}

// ---------------------------------------------------------------------------
// Prep: w_hh -> bf16, bsum = b_ih + b_hh
// ---------------------------------------------------------------------------
__global__ void prep_kernel(const float* __restrict__ w_hh,
                            const float* __restrict__ b_ih,
                            const float* __restrict__ b_hh,
                            unsigned short* __restrict__ whh_bf,
                            float* __restrict__ bsum) {
  int idx = blockIdx.x * blockDim.x + threadIdx.x;
  if (idx < FOURH * H) whh_bf[idx] = f2bf(w_hh[idx]);
  if (idx < FOURH)     bsum[idx]   = b_ih[idx] + b_hh[idx];
}

// ---------------------------------------------------------------------------
// LSTM: 8 blocks x 16 batch rows, 1024 sequential steps.
// gates(16x1024) = xt*wi + h @ w_hh^T + b via bf16 WMMA.
// First two K-slices of B (w_hh) are held in registers across all timesteps
// so the post-barrier WMMAs have no load dependence (hides L2 latency of the
// 6 streamed slices). h in LDS (bf16), c in registers.
// ---------------------------------------------------------------------------
__global__ __launch_bounds__(256) void lstm_kernel(
    const float* __restrict__ x,
    const float* __restrict__ w_ih,            // (4H,1) -> vector wi
    const unsigned short* __restrict__ whh_bf, // (4H,H) bf16 row-major
    const float* __restrict__ bsum,            // 4H
    unsigned short* __restrict__ hs_out)       // (B, INS*H) bf16
{
  __shared__ float          gates[16 * FOURH];  // 64 KB
  __shared__ unsigned short hlds[16 * H];       // 8 KB
  __shared__ float          xts[16];

  const int tid  = threadIdx.x;
  const int lane = tid & 31;
  const int wv   = tid >> 5;         // wave 0..7
  const int m0   = blockIdx.x * 16;  // batch-row base

  const int hl = lane >> 4;          // lane half
  const int ln = lane & 15;
  const int colBase = wv * 128;

  // per-lane wi / bias for this wave's 8 N-tiles
  float wiR[8], bsR[8];
#pragma unroll
  for (int nt = 0; nt < 8; ++nt) {
    int col = colBase + nt * 16 + ln;
    wiR[nt] = w_ih[col];
    bsR[nt] = bsum[col];
  }

  // resident B fragments for K-slices kb=0 and kb=32 (invariant across t)
  FragBF bRes[2][8];
#pragma unroll
  for (int s = 0; s < 2; ++s) {
#pragma unroll
    for (int nt = 0; nt < 8; ++nt) {
      int col = colBase + nt * 16 + ln;
      const uint4* bp4 = (const uint4*)&whh_bf[(size_t)col * H + s * 32 + hl * 16];
      uint4 q0 = bp4[0], q1 = bp4[1];
      bRes[s][nt].u[0] = q0.x; bRes[s][nt].u[1] = q0.y;
      bRes[s][nt].u[2] = q0.z; bRes[s][nt].u[3] = q0.w;
      bRes[s][nt].u[4] = q1.x; bRes[s][nt].u[5] = q1.y;
      bRes[s][nt].u[6] = q1.z; bRes[s][nt].u[7] = q1.w;
    }
  }

  // h0 = 0
  for (int i = tid; i < 16 * H; i += 256) hlds[i] = 0;

  // elementwise ownership: row em, cols jb..jb+15 ; c state in registers
  const int em = tid >> 4;
  const int jb = (tid & 15) * 16;
  float creg[16];
#pragma unroll
  for (int e = 0; e < 16; ++e) creg[e] = 0.0f;

  for (int t = 0; t < T_STEPS; ++t) {
    if (tid < 16) {
      int xi = (t % INS) * CHUNKS + (t / INS);   // xseq permutation
      xts[tid] = x[(size_t)(m0 + tid) * (CHUNKS * INS) + xi];
    }
    __syncthreads();

    // accumulator init: xt[b]*wi[n] + bsum[n]   (C layout: row = v+8*hl, col = ln)
    float xr[8];
#pragma unroll
    for (int v = 0; v < 8; ++v) xr[v] = xts[v + 8 * hl];
    v8f acc[8];
#pragma unroll
    for (int nt = 0; nt < 8; ++nt) {
#pragma unroll
      for (int v = 0; v < 8; ++v) acc[nt][v] = xr[v] * wiR[nt] + bsR[nt];
    }

    // K-slices 0,32: B resident in registers -> immediate WMMA issue
#pragma unroll
    for (int s = 0; s < 2; ++s) {
      int kb = s * 32;
      FragBF a;
#pragma unroll
      for (int j = 0; j < 8; ++j) {
        int kk = hl * 16 + 2 * j;
        a.u[j] = *(const unsigned*)&hlds[ln * H + kb + kk];
      }
#pragma unroll
      for (int nt = 0; nt < 8; ++nt) {
        acc[nt] = __builtin_amdgcn_wmma_f32_16x16x32_bf16(
            false, a.v, false, bRes[s][nt].v, (short)0, acc[nt], false, false);
      }
    }

    // K-slices 64..224: stream B from L2
    for (int kb = 64; kb < H; kb += 32) {
      FragBF a;
#pragma unroll
      for (int j = 0; j < 8; ++j) {
        int kk = hl * 16 + 2 * j;
        a.u[j] = *(const unsigned*)&hlds[ln * H + kb + kk];
      }
#pragma unroll
      for (int nt = 0; nt < 8; ++nt) {
        int col = colBase + nt * 16 + ln;
        const uint4* bp4 = (const uint4*)&whh_bf[(size_t)col * H + kb + hl * 16];
        uint4 q0 = bp4[0], q1 = bp4[1];
        FragBF b;
        b.u[0] = q0.x; b.u[1] = q0.y; b.u[2] = q0.z; b.u[3] = q0.w;
        b.u[4] = q1.x; b.u[5] = q1.y; b.u[6] = q1.z; b.u[7] = q1.w;
        acc[nt] = __builtin_amdgcn_wmma_f32_16x16x32_bf16(
            false, a.v, false, b.v, (short)0, acc[nt], false, false);
      }
    }

    // gates to LDS
#pragma unroll
    for (int nt = 0; nt < 8; ++nt) {
      int col = colBase + nt * 16 + ln;
#pragma unroll
      for (int v = 0; v < 8; ++v) gates[(v + 8 * hl) * FOURH + col] = acc[nt][v];
    }
    __syncthreads();

    // LSTM cell update
    unsigned short hrow[16];
#pragma unroll
    for (int e = 0; e < 16; ++e) {
      int j = jb + e;
      float ig = fast_sigmoid(gates[em * FOURH + j]);
      float fg = fast_sigmoid(gates[em * FOURH + j + H]);
      float gg = fast_tanh(gates[em * FOURH + j + 2 * H]);
      float og = fast_sigmoid(gates[em * FOURH + j + 3 * H]);
      float c = fg * creg[e] + ig * gg;
      creg[e] = c;
      float h = og * fast_tanh(c);
      unsigned short hb = f2bf(h);
      hlds[em * H + j] = hb;
      hrow[e] = hb;
    }
    if (t >= INS) {
      uint4 p0, p1;
      p0.x = (unsigned)hrow[0]  | ((unsigned)hrow[1]  << 16);
      p0.y = (unsigned)hrow[2]  | ((unsigned)hrow[3]  << 16);
      p0.z = (unsigned)hrow[4]  | ((unsigned)hrow[5]  << 16);
      p0.w = (unsigned)hrow[6]  | ((unsigned)hrow[7]  << 16);
      p1.x = (unsigned)hrow[8]  | ((unsigned)hrow[9]  << 16);
      p1.y = (unsigned)hrow[10] | ((unsigned)hrow[11] << 16);
      p1.z = (unsigned)hrow[12] | ((unsigned)hrow[13] << 16);
      p1.w = (unsigned)hrow[14] | ((unsigned)hrow[15] << 16);
      uint4* dst = (uint4*)&hs_out[(size_t)(m0 + em) * KBIG +
                                   (size_t)(t - INS) * H + jb];
      dst[0] = p0; dst[1] = p1;
    }
    __syncthreads();
  }
}

// ---------------------------------------------------------------------------
// Big GEMM: res1 = relu(hs(128x131072,bf16) @ w1^T + b1).
// 16 blocks (one N-tile each); every block covers all 8 M-tiles so each
// converted B-fragment (w1 f32 -> bf16 in registers) is reused 8x.
// 8 waves split K; LDS reduction + bias + relu.
// ---------------------------------------------------------------------------
__global__ __launch_bounds__(256) void gemm1_kernel(
    const unsigned short* __restrict__ hsA,  // (128, KBIG) bf16
    const float* __restrict__ w1,            // (H, KBIG) f32
    const float* __restrict__ b1,
    float* __restrict__ res1)                // (128, H) f32
{
  const int nt  = blockIdx.x;   // 0..15
  const int tid = threadIdx.x;
  const int lane = tid & 31;
  const int wv   = tid >> 5;
  const int hl   = lane >> 4;
  const int ln   = lane & 15;

  __shared__ float red[8][8][16][16];   // [wave][mt][m][n] 64KB

  const int kSlice = KBIG / 8;          // 16384
  const int k0 = wv * kSlice;

  const float* brow = w1 + (size_t)(nt * 16 + ln) * KBIG;

  v8f acc[8] = {};
  for (int kb = k0; kb < k0 + kSlice; kb += 32) {
    // B fragment: 16 f32 of w1 row -> bf16, built once, used for 8 M-tiles
    const float4* bp = (const float4*)&brow[kb + hl * 16];
    float4 f0 = bp[0], f1 = bp[1], f2 = bp[2], f3 = bp[3];
    FragBF b;
    b.u[0] = pack2bf(f0.x, f0.y); b.u[1] = pack2bf(f0.z, f0.w);
    b.u[2] = pack2bf(f1.x, f1.y); b.u[3] = pack2bf(f1.z, f1.w);
    b.u[4] = pack2bf(f2.x, f2.y); b.u[5] = pack2bf(f2.z, f2.w);
    b.u[6] = pack2bf(f3.x, f3.y); b.u[7] = pack2bf(f3.z, f3.w);
    __builtin_prefetch(&brow[kb + 2048], 0, 0);   // global_prefetch_b8 on w1 stream
#pragma unroll
    for (int mt = 0; mt < 8; ++mt) {
      const unsigned short* arow = hsA + (size_t)(mt * 16 + ln) * KBIG;
      FragBF a;
#pragma unroll
      for (int j = 0; j < 8; ++j) {
        int kk = hl * 16 + 2 * j;
        a.u[j] = *(const unsigned*)&arow[kb + kk];
      }
      acc[mt] = __builtin_amdgcn_wmma_f32_16x16x32_bf16(
          false, a.v, false, b.v, (short)0, acc[mt], false, false);
    }
  }
#pragma unroll
  for (int mt = 0; mt < 8; ++mt)
#pragma unroll
    for (int v = 0; v < 8; ++v) red[wv][mt][v + 8 * hl][ln] = acc[mt][v];
  __syncthreads();

  // reduce 8 waves -> 2048 output cells (8 per thread)
  for (int c = tid; c < 2048; c += 256) {
    int mt = c >> 8, m = (c >> 4) & 15, n = c & 15;
    float s = 0.0f;
#pragma unroll
    for (int w = 0; w < 8; ++w) s += red[w][mt][m][n];
    s += b1[nt * 16 + n];
    s = fmaxf(s, 0.0f);
    res1[(size_t)(mt * 16 + m) * H + nt * 16 + n] = s;
  }
}

// ---------------------------------------------------------------------------
// Heads: res2 = res1@w2^T+b2 ; out = res2@w3^T+b3. One block per batch row.
// ---------------------------------------------------------------------------
__global__ __launch_bounds__(256) void head_kernel(
    const float* __restrict__ res1,  // (128, H)
    const float* __restrict__ w2, const float* __restrict__ b2,
    const float* __restrict__ w3, const float* __restrict__ b3,
    float* __restrict__ out)         // (128, OS)
{
  __shared__ float r1[H];
  __shared__ float r2[H2];
  const int b = blockIdx.x;
  const int tid = threadIdx.x;
  r1[tid] = res1[(size_t)b * H + tid];
  __syncthreads();
#pragma unroll
  for (int rep = 0; rep < 2; ++rep) {
    int n = tid + rep * 256;
    const float* wrow = w2 + (size_t)n * H;
    float s = b2[n];
    for (int k = 0; k < H; ++k) s += r1[k] * wrow[k];
    r2[n] = s;
  }
  __syncthreads();
  if (tid < OS) {
    const float* wrow = w3 + (size_t)tid * H2;
    float s = b3[tid];
    for (int k = 0; k < H2; ++k) s += r2[k] * wrow[k];
    out[(size_t)b * OS + tid] = s;
  }
}

// ---------------------------------------------------------------------------
extern "C" void kernel_launch(void* const* d_in, const int* in_sizes, int n_in,
                              void* d_out, int out_size, void* d_ws, size_t ws_size,
                              hipStream_t stream) {
  const float* x    = (const float*)d_in[0];
  const float* w_ih = (const float*)d_in[1];
  const float* w_hh = (const float*)d_in[2];
  const float* b_ih = (const float*)d_in[3];
  const float* b_hh = (const float*)d_in[4];
  const float* w1   = (const float*)d_in[5];
  const float* b1   = (const float*)d_in[6];
  const float* w2   = (const float*)d_in[7];
  const float* b2   = (const float*)d_in[8];
  const float* w3   = (const float*)d_in[9];
  const float* b3   = (const float*)d_in[10];

  char* ws = (char*)d_ws;
  unsigned short* whh_bf = (unsigned short*)ws;              // 512 KB
  float*          bsum   = (float*)(ws + 524288);            // 4 KB
  unsigned short* hs     = (unsigned short*)(ws + 528384);   // 32 MB
  float*          res1   = (float*)(ws + 528384 + 33554432); // 128 KB

  prep_kernel<<<dim3((FOURH * H + 255) / 256), 256, 0, stream>>>(
      w_hh, b_ih, b_hh, whh_bf, bsum);
  lstm_kernel<<<dim3(8), 256, 0, stream>>>(x, w_ih, whh_bf, bsum, hs);
  gemm1_kernel<<<dim3(16), 256, 0, stream>>>(hs, w1, b1, res1);
  head_kernel<<<dim3(B_SZ), 256, 0, stream>>>(res1, w2, b2, w3, b3, (float*)d_out);
}
